// net_5617817223717
// MI455X (gfx1250) — matrix operand order repaired and checked by reading, not compiled, not run on hardware
//
#include <hip/hip_runtime.h>

typedef _Float16 h16;
typedef __attribute__((ext_vector_type(16))) _Float16 v16h;
typedef __attribute__((ext_vector_type(8)))  _Float16 v8h;
typedef __attribute__((ext_vector_type(8)))  float    v8f;

#define Bn   2
#define Cn   128
#define Hn   64
#define Wn   64
#define Ln   (Hn * Wn)       // 4096
#define Mn   (Bn * Ln)       // 8192 tokens
#define DIn  256
#define DSn  16
#define DRn  8
#define HIDn 256
#define NPXP 48              // x_proj N padded 40 -> 48

#define CDIV(a, b) (((a) + (b) - 1) / (b))

// Async global->LDS path (gfx1250): guard on builtin availability; fall back to
// a synchronous LDS staging copy when the toolchain lacks the builtins.
#if defined(__HIP_DEVICE_COMPILE__) && defined(__gfx1250__) && \
    __has_builtin(__builtin_amdgcn_global_load_async_to_lds_b128) && \
    __has_builtin(__builtin_amdgcn_s_wait_asynccnt)
#define USE_ASYNC_LDS 1
// Exact parameter types per hipcc diagnostic: non-const pointer to a 16-byte
// int vector (GCC vector_size style), global addrspace(1) src / LDS addrspace(3) dst.
typedef int v4i_vs __attribute__((vector_size(4 * sizeof(int))));
typedef __attribute__((address_space(1))) v4i_vs* async_gptr_t;
typedef __attribute__((address_space(3))) v4i_vs* async_lptr_t;
#else
#define USE_ASYNC_LDS 0
#endif

// ---------------- WMMA fragment helpers (ISA 7.12.2 layouts, wave32) ----------

__device__ __forceinline__ v16h frag_a(const h16* __restrict__ tile, int lda, int lane) {
  // A 16x32 f16: lane&15 = row M; lanes<16: K={0..7,16..23}; lanes>=16: K={8..15,24..31}
  const h16* q = tile + (size_t)(lane & 15) * lda + ((lane >> 4) << 3);
  v8h lo = *(const v8h*)q;
  v8h hi = *(const v8h*)(q + 16);
  return __builtin_shufflevector(lo, hi, 0,1,2,3,4,5,6,7,8,9,10,11,12,13,14,15);
}

__device__ __forceinline__ v16h frag_b(const h16* __restrict__ wtile, int ldw, int lane) {
  // B 32x16: lane&15 = col N; lanes<16: K={0..7,16..23}; lanes>=16: K={8..15,24..31}
  const h16* q = wtile + (size_t)(lane & 15) * ldw + ((lane >> 4) << 3);
  v8h lo = *(const v8h*)q;
  v8h hi = *(const v8h*)(q + 16);
  return __builtin_shufflevector(lo, hi, 0,1,2,3,4,5,6,7,8,9,10,11,12,13,14,15);
}

// ---------------- Generic GEMM: C(M,Npad) = A(M,KT) x W(Npad,KT)^T ------------
// block = 128 threads (4 waves); each wave computes a 16 x (16*NSUB) tile.
// The block's (16*NSUB x KT) fp16 weight panel is staged into LDS once
// (async global->LDS on CDNA5), then B fragments are fed from LDS.
template<int NSUB, int KT>
__global__ void gemm_wmma_kernel(const h16* __restrict__ A, const h16* __restrict__ W,
                                 float* __restrict__ C, int Npad) {
  __shared__ alignas(32) h16 wlds[16 * NSUB * KT];
  const int lane = threadIdx.x & 31;
  const int wave = threadIdx.x >> 5;
  const int m0 = (blockIdx.x * 4 + wave) * 16;          // caller guarantees m0 < M
  const int n0 = blockIdx.y * (16 * NSUB);

  const h16* wsrc = W + (size_t)n0 * KT;
  for (int i = threadIdx.x * 8; i < 16 * NSUB * KT; i += 128 * 8) {
#if USE_ASYNC_LDS
    __builtin_amdgcn_global_load_async_to_lds_b128(
        (async_gptr_t)(wsrc + i), (async_lptr_t)(wlds + i), 0, 0);
#else
    *(v8h*)(wlds + i) = *(const v8h*)(wsrc + i);
#endif
  }
#if USE_ASYNC_LDS
  __builtin_amdgcn_s_wait_asynccnt(0);
#endif
  __syncthreads();

  v8f zero = {};
  v8f acc[NSUB];
#pragma unroll
  for (int t = 0; t < NSUB; ++t) acc[t] = zero;
  const h16* arow = A + (size_t)m0 * KT;
#pragma unroll
  for (int k0 = 0; k0 < KT; k0 += 32) {
    if (k0 + 32 < KT) __builtin_prefetch(arow + k0 + 32, 0, 3);  // global_prefetch_b8
    v16h a = frag_a(arow + k0, KT, lane);
#pragma unroll
    for (int t = 0; t < NSUB; ++t) {
      v16h b = frag_b(wlds + (size_t)(16 * t) * KT + k0, KT, lane);   // ds_load path
      acc[t] = __builtin_amdgcn_wmma_f32_16x16x32_f16(false, a, false, b,
                                                      (short)0, acc[t], false, false);
    }
  }
  const int mb = m0 + ((lane >> 4) << 3);
#pragma unroll
  for (int t = 0; t < NSUB; ++t) {
    float* p = C + (size_t)mb * Npad + n0 + 16 * t + (lane & 15);
#pragma unroll
    for (int i = 0; i < 8; ++i) p[(size_t)i * Npad] = acc[t][i];
  }
}

// ---------------- 3x3 conv (pad 1) as implicit GEMM over 9 taps ---------------
// X: token-major (B*H*W, Cn) f16; Wt: [tap][co][ci] f16; Out: (B*H*W, Cn) f32 (+bias)
__global__ void conv3x3_wmma_kernel(const h16* __restrict__ X, const h16* __restrict__ Wt,
                                    const float* __restrict__ bias, float* __restrict__ Out) {
  const int lane = threadIdx.x & 31;
  const int wave = threadIdx.x >> 5;
  const int tile = blockIdx.x * 4 + wave;          // B*H*(W/16) = 512 tiles
  const int b  = tile >> 8;                        // 256 tiles per batch
  const int rem = tile & 255;
  const int y  = rem >> 2;
  const int x0 = (rem & 3) << 4;
  const int n0 = blockIdx.y * 64;
  const int m  = lane & 15;
  const int kh = (lane >> 4) << 3;
  v8f zero = {};
  v8f acc[4];
#pragma unroll
  for (int t = 0; t < 4; ++t) acc[t] = zero;
  for (int ky = 0; ky < 3; ++ky) {
    const int yy = y + ky - 1;
    for (int kx = 0; kx < 3; ++kx) {
      const int xx = x0 + m + kx - 1;
      const bool ok = (yy >= 0) && (yy < Hn) && (xx >= 0) && (xx < Wn);
      const h16* abase = X + (((size_t)b * Hn + (ok ? yy : 0)) * Wn + (ok ? xx : 0)) * Cn;
      const h16* wbase = Wt + (size_t)(ky * 3 + kx) * Cn * Cn;
#pragma unroll
      for (int k0 = 0; k0 < Cn; k0 += 32) {
        v16h a = {};
        if (ok) {
          const h16* q = abase + k0 + kh;
          v8h lo = *(const v8h*)q;
          v8h hi = *(const v8h*)(q + 16);
          a = __builtin_shufflevector(lo, hi, 0,1,2,3,4,5,6,7,8,9,10,11,12,13,14,15);
        }
#pragma unroll
        for (int t = 0; t < 4; ++t) {
          v16h bf = frag_b(wbase + (size_t)(n0 + 16 * t) * Cn + k0, Cn, lane);
          acc[t] = __builtin_amdgcn_wmma_f32_16x16x32_f16(false, a, false, bf,
                                                          (short)0, acc[t], false, false);
        }
      }
    }
  }
  const size_t row0 = ((size_t)b * Hn + y) * Wn + x0 + ((lane >> 4) << 3);
#pragma unroll
  for (int t = 0; t < 4; ++t) {
    const int col = n0 + 16 * t + (lane & 15);
    const float bv = bias[col];
    float* p = Out + row0 * Cn + col;
#pragma unroll
    for (int i = 0; i < 8; ++i) p[(size_t)i * Cn] = acc[t][i] + bv;
  }
}

// ---------------- Elementwise / scan kernels ----------------------------------

__device__ __forceinline__ float siluf(float v) { return v / (1.f + expf(-v)); }

__global__ void f32_to_f16_kernel(const float* __restrict__ s, h16* __restrict__ d, int n) {
  int t = blockIdx.x * blockDim.x + threadIdx.x;
  if (t < n) d[t] = (h16)s[t];
}

__global__ void pad_xpw_kernel(const float* __restrict__ s, h16* __restrict__ d) {
  int t = blockIdx.x * blockDim.x + threadIdx.x;     // 48*256
  if (t >= NPXP * DIn) return;
  int n = t >> 8, k = t & 255;
  d[t] = (n < DRn + 2 * DSn) ? (h16)s[n * DIn + k] : (h16)0.f;
}

__global__ void repack_conv_kernel(const float* __restrict__ s, h16* __restrict__ d) {
  int t = blockIdx.x * blockDim.x + threadIdx.x;     // 9*128*128
  if (t >= 9 * Cn * Cn) return;
  int ci = t & 127, co = (t >> 7) & 127, tap = t >> 14;
  int ky = tap / 3, kx = tap % 3;
  d[t] = (h16)s[((size_t)(co * Cn + ci) * 3 + ky) * 3 + kx];
}

// channels_first LN over C of x(B,C,H,W); writes token-major h(M,C) f32 and raw x as f16
__global__ void ln_cf_kernel(const float* __restrict__ x, const float* __restrict__ w,
                             const float* __restrict__ b, float* __restrict__ h,
                             h16* __restrict__ xh) {
  int p = blockIdx.x * blockDim.x + threadIdx.x;
  if (p >= Mn) return;
  int bb = p >> 12, sp = p & (Ln - 1);
  const float* xp = x + (size_t)bb * Cn * Ln + sp;
  float sum = 0.f, sq = 0.f;
  for (int c = 0; c < Cn; ++c) { float v = xp[(size_t)c * Ln]; sum += v; sq += v * v; }
  float mu = sum * (1.f / Cn);
  float var = sq * (1.f / Cn) - mu * mu;
  float inv = rsqrtf(var + 1e-6f);
  float* hp = h + (size_t)p * Cn;
  h16* xo = xh + (size_t)p * Cn;
  for (int c = 0; c < Cn; ++c) {
    float v = xp[(size_t)c * Ln];
    hp[c] = w[c] * ((v - mu) * inv) + b[c];
    xo[c] = (h16)v;
  }
}

// rowwise LN over contiguous last dim of (M,C) f32 -> f16
__global__ void ln_row_kernel(const float* __restrict__ in, const float* __restrict__ w,
                              const float* __restrict__ b, h16* __restrict__ out, float eps) {
  int r = blockIdx.x * blockDim.x + threadIdx.x;
  if (r >= Mn) return;
  const float* p = in + (size_t)r * Cn;
  float sum = 0.f, sq = 0.f;
  for (int c = 0; c < Cn; ++c) { float v = p[c]; sum += v; sq += v * v; }
  float mu = sum * (1.f / Cn);
  float var = sq * (1.f / Cn) - mu * mu;
  float inv = rsqrtf(var + eps);
  h16* o = out + (size_t)r * Cn;
  for (int c = 0; c < Cn; ++c) o[c] = (h16)((p[c] - mu) * inv * w[c] + b[c]);
}

// causal depthwise conv1d (DC=4) + SiLU; rev=1 gathers from reversed token order
__global__ void dwconv_silu_kernel(const float* __restrict__ xz, const float* __restrict__ cw,
                                   const float* __restrict__ cb, h16* __restrict__ uc, int rev) {
  int t = blockIdx.x * blockDim.x + threadIdx.x;
  if (t >= Mn * DIn) return;
  int d = t & (DIn - 1); int row = t >> 8;
  int b = row >> 12; int l = row & (Ln - 1);
  float acc = cb[d];
#pragma unroll
  for (int j = 0; j < 4; ++j) {
    int ls = l - 3 + j;
    if (ls < 0) continue;
    int src = rev ? (Ln - 1 - ls) : ls;
    acc += xz[((size_t)(b * Ln + src)) * (2 * DIn) + d] * cw[d * 4 + j];
  }
  uc[t] = (h16)siluf(acc);
}

// delta = softplus(dt @ dtW^T + dtB); dt = dbl cols 0..7
__global__ void delta_kernel(const float* __restrict__ dbl, const float* __restrict__ dtW,
                             const float* __restrict__ dtB, float* __restrict__ delta) {
  int t = blockIdx.x * blockDim.x + threadIdx.x;
  if (t >= Mn * DIn) return;
  int d = t & (DIn - 1); int row = t >> 8;
  const float* db = dbl + (size_t)row * NPXP;
  float acc = dtB[d];
#pragma unroll
  for (int r = 0; r < DRn; ++r) acc += db[r] * dtW[d * DRn + r];
  delta[t] = (acc > 20.f) ? acc : log1pf(expf(acc));
}

// selective scan: one thread per (b,d,s); 16-state reduce via shfl_xor in 16-lane groups
__global__ void scan_kernel(const float* __restrict__ delta, const h16* __restrict__ uc,
                            const float* __restrict__ dbl, const float* __restrict__ Alog,
                            float* __restrict__ ys) {
  int tid = blockIdx.x * blockDim.x + threadIdx.x;   // Bn*DIn*DSn = 8192
  int s = tid & 15;
  int d = (tid >> 4) & (DIn - 1);
  int b = tid >> 12;
  float A = -expf(Alog[d * DSn + s]);
  float h = 0.f;
  const float* dl = delta + (size_t)b * Ln * DIn + d;
  const h16*  up = uc    + (size_t)b * Ln * DIn + d;
  const float* bp = dbl  + (size_t)b * Ln * NPXP + DRn + s;
  const float* cp = dbl  + (size_t)b * Ln * NPXP + DRn + DSn + s;
  float* yp = ys + (size_t)b * Ln * DIn + d;
  for (int l = 0; l < Ln; ++l) {
    float dt = dl[(size_t)l * DIn];
    float u  = (float)up[(size_t)l * DIn];
    float Bv = bp[(size_t)l * NPXP];
    float Cv = cp[(size_t)l * NPXP];
    h = expf(dt * A) * h + dt * u * Bv;
    float y = h * Cv;
    y += __shfl_xor(y, 1, 32);
    y += __shfl_xor(y, 2, 32);
    y += __shfl_xor(y, 4, 32);
    y += __shfl_xor(y, 8, 32);
    if (s == 0) yp[(size_t)l * DIn] = y;
  }
}

// ycomb[l] = gate_f(yfull_f[l]) + gate_b(yfull_b[L-1-l])  (un-reverses backward dir)
__global__ void combine_kernel(const float* __restrict__ ys_f, const h16* __restrict__ uc_f,
                               const float* __restrict__ xz_f, const float* __restrict__ Dp_f,
                               const float* __restrict__ ys_b, const h16* __restrict__ uc_b,
                               const float* __restrict__ xz_b, const float* __restrict__ Dp_b,
                               h16* __restrict__ yc) {
  int t = blockIdx.x * blockDim.x + threadIdx.x;
  if (t >= Mn * DIn) return;
  int d = t & (DIn - 1); int row = t >> 8;
  int b = row >> 12; int l = row & (Ln - 1);
  float zf = xz_f[(size_t)row * (2 * DIn) + DIn + d];
  float yf = (ys_f[t] + (float)uc_f[t] * Dp_f[d]) * siluf(zf);
  size_t tb = ((size_t)(b * Ln + (Ln - 1 - l))) * DIn + d;
  float zb = xz_b[(size_t)row * (2 * DIn) + DIn + d];   // z of same token (reversed scan idx)
  float yb = (ys_b[tb] + (float)uc_b[tb] * Dp_b[d]) * siluf(zb);
  yc[t] = (h16)(yf + yb);
}

__global__ void residual1_kernel(const float* __restrict__ x, const float* __restrict__ bm,
                                 const float* __restrict__ h, const float* __restrict__ g1,
                                 float* __restrict__ xa) {
  int t = blockIdx.x * blockDim.x + threadIdx.x;
  if (t >= Mn * Cn) return;
  int c = t & (Cn - 1); int row = t >> 7;
  int b = row >> 12; int sp = row & (Ln - 1);
  float xv = x[(size_t)b * Cn * Ln + (size_t)c * Ln + sp];
  xa[t] = xv + g1[c] * (bm[t] + h[t]);
}

__global__ void residual2_kernel(const float* __restrict__ xa, const float* __restrict__ gms,
                                 const float* __restrict__ g2, float* __restrict__ x2a) {
  int t = blockIdx.x * blockDim.x + threadIdx.x;
  if (t >= Mn * Cn) return;
  int c = t & (Cn - 1);
  x2a[t] = xa[t] + g2[c] * gms[t];
}

// MSFF gating: three depthwise 3x3 convs (dil 1,2,3) on ph(M,768) -> gelu(x1)*x2*x3
__global__ void msff_kernel(const float* __restrict__ ph, const float* __restrict__ dw1,
                            const float* __restrict__ dw2, const float* __restrict__ dw3,
                            h16* __restrict__ g) {
  int t = blockIdx.x * blockDim.x + threadIdx.x;
  if (t >= Mn * HIDn) return;
  int c = t & (HIDn - 1); int pix = t >> 8;
  int b = pix >> 12; int sp = pix & (Ln - 1);
  int y = sp >> 6, xx = sp & 63;
  const float* wts[3] = { dw1, dw2, dw3 };
  float r[3];
#pragma unroll
  for (int i = 0; i < 3; ++i) {
    int dil = i + 1;
    const float* w = wts[i] + c * 9;
    float acc = 0.f;
#pragma unroll
    for (int ky = 0; ky < 3; ++ky) {
#pragma unroll
      for (int kx = 0; kx < 3; ++kx) {
        int yy = y + (ky - 1) * dil, xc = xx + (kx - 1) * dil;
        if (yy >= 0 && yy < Hn && xc >= 0 && xc < Wn)
          acc += ph[((size_t)(b * Ln + yy * Wn + xc)) * (3 * HIDn) + i * HIDn + c] * w[ky * 3 + kx];
      }
    }
    r[i] = acc;
  }
  float ge = 0.5f * r[0] * (1.f + erff(r[0] * 0.70710678118f));
  g[t] = (h16)(ge * r[1] * r[2]);
}

__global__ void bnrelu_kernel(const float* __restrict__ in, const float* __restrict__ w,
                              const float* __restrict__ b, const float* __restrict__ m,
                              const float* __restrict__ v, h16* __restrict__ oh,
                              float* __restrict__ of) {
  int t = blockIdx.x * blockDim.x + threadIdx.x;
  if (t >= Mn * Cn) return;
  int c = t & (Cn - 1);
  float val = (in[t] - m[c]) * rsqrtf(v[c] + 1e-5f) * w[c] + b[c];
  val = fmaxf(val, 0.f);
  oh[t] = (h16)val;
  of[t] = val;
}

__global__ void final_kernel(const float* __restrict__ x2a, const float* __restrict__ f,
                             float* __restrict__ out) {
  int t = blockIdx.x * blockDim.x + threadIdx.x;    // B*C*H*W layout
  if (t >= Bn * Cn * Ln) return;
  int b = t / (Cn * Ln);
  int r = t - b * Cn * Ln;
  int c = r >> 12;
  int sp = r & (Ln - 1);
  size_t row = (size_t)b * Ln + sp;
  out[t] = x2a[row * Cn + c] + f[row * Cn + c];
}

// ---------------- host-side orchestration ------------------------------------

extern "C" void kernel_launch(void* const* d_in, const int* in_sizes, int n_in,
                              void* d_out, int out_size, void* d_ws, size_t ws_size,
                              hipStream_t stream) {
  (void)in_sizes; (void)n_in; (void)out_size; (void)ws_size;
  const float* x      = (const float*)d_in[0];
  const float* ln1_w  = (const float*)d_in[1];
  const float* ln1_b  = (const float*)d_in[2];
  const float* bmw    = (const float*)d_in[3];
  const float* bmb    = (const float*)d_in[4];
  const float* inW[2]   = { (const float*)d_in[5],  (const float*)d_in[13] };
  const float* convW[2] = { (const float*)d_in[6],  (const float*)d_in[14] };
  const float* convB[2] = { (const float*)d_in[7],  (const float*)d_in[15] };
  const float* xpW[2]   = { (const float*)d_in[8],  (const float*)d_in[16] };
  const float* dtW[2]   = { (const float*)d_in[9],  (const float*)d_in[17] };
  const float* dtB[2]   = { (const float*)d_in[10], (const float*)d_in[18] };
  const float* Alog[2]  = { (const float*)d_in[11], (const float*)d_in[19] };
  const float* Dp[2]    = { (const float*)d_in[12], (const float*)d_in[20] };
  const float* outW   = (const float*)d_in[21];
  const float* gamma1 = (const float*)d_in[22];
  const float* gamma2 = (const float*)d_in[23];
  const float* msin   = (const float*)d_in[24];
  const float* dw1    = (const float*)d_in[25];
  const float* dw2    = (const float*)d_in[26];
  const float* dw3    = (const float*)d_in[27];
  const float* msout  = (const float*)d_in[28];
  const float* c1W = (const float*)d_in[29]; const float* c1B = (const float*)d_in[30];
  const float* bn1w = (const float*)d_in[31]; const float* bn1b = (const float*)d_in[32];
  const float* bn1m = (const float*)d_in[33]; const float* bn1v = (const float*)d_in[34];
  const float* c2W = (const float*)d_in[35]; const float* c2B = (const float*)d_in[36];
  const float* bn2w = (const float*)d_in[37]; const float* bn2b = (const float*)d_in[38];
  const float* bn2m = (const float*)d_in[39]; const float* bn2v = (const float*)d_in[40];
  float* out = (float*)d_out;

  char* ws = (char*)d_ws;
  size_t off = 0;
  auto alloc = [&](size_t bytes) -> void* {
    void* p = ws + off; off += (bytes + 255) & ~(size_t)255; return p;
  };

  // persistent
  h16*   xn_h  = (h16*)  alloc((size_t)Mn * Cn * 2);
  float* h32   = (float*)alloc((size_t)Mn * Cn * 4);
  h16*   xh    = (h16*)  alloc((size_t)Mn * Cn * 2);
  h16*   ycomb = (h16*)  alloc((size_t)Mn * DIn * 2);
  float* bm32  = (float*)alloc((size_t)Mn * Cn * 4);
  float* xa32  = (float*)alloc((size_t)Mn * Cn * 4);
  h16*   ln2h  = (h16*)  alloc((size_t)Mn * Cn * 2);
  float* x2a   = (float*)alloc((size_t)Mn * Cn * 4);
  h16* w_in0  = (h16*)alloc((size_t)2 * DIn * Cn * 2);
  h16* w_in1  = (h16*)alloc((size_t)2 * DIn * Cn * 2);
  h16* w_xp0  = (h16*)alloc((size_t)NPXP * DIn * 2);
  h16* w_xp1  = (h16*)alloc((size_t)NPXP * DIn * 2);
  h16* w_out  = (h16*)alloc((size_t)Cn * DIn * 2);
  h16* w_msin = (h16*)alloc((size_t)3 * HIDn * Cn * 2);
  h16* w_msout= (h16*)alloc((size_t)Cn * HIDn * 2);
  h16* w_c1   = (h16*)alloc((size_t)9 * Cn * Cn * 2);
  h16* w_c2   = (h16*)alloc((size_t)9 * Cn * Cn * 2);
  size_t shared = off;

  // phase 1 transients (mamba)
  float* xz0 = (float*)alloc((size_t)Mn * 2 * DIn * 4);
  float* xz1 = (float*)alloc((size_t)Mn * 2 * DIn * 4);
  h16*   uc0 = (h16*)  alloc((size_t)Mn * DIn * 2);
  h16*   uc1 = (h16*)  alloc((size_t)Mn * DIn * 2);
  float* db0 = (float*)alloc((size_t)Mn * NPXP * 4);
  float* db1 = (float*)alloc((size_t)Mn * NPXP * 4);
  float* de0 = (float*)alloc((size_t)Mn * DIn * 4);
  float* de1 = (float*)alloc((size_t)Mn * DIn * 4);
  float* ys0 = (float*)alloc((size_t)Mn * DIn * 4);
  float* ys1 = (float*)alloc((size_t)Mn * DIn * 4);

  // phase 2 transients (MSFF + fusion) reuse phase-1 region
  off = shared;
  float* ph  = (float*)alloc((size_t)Mn * 3 * HIDn * 4);
  h16*   g_h = (h16*)  alloc((size_t)Mn * HIDn * 2);
  float* gms = (float*)alloc((size_t)Mn * Cn * 4);
  float* c1o = (float*)alloc((size_t)Mn * Cn * 4);
  h16*   f1h = (h16*)  alloc((size_t)Mn * Cn * 2);
  float* f1f = (float*)alloc((size_t)Mn * Cn * 4);
  float* c2o = (float*)alloc((size_t)Mn * Cn * 4);
  h16*   f2h = (h16*)  alloc((size_t)Mn * Cn * 2);
  float* f2f = (float*)alloc((size_t)Mn * Cn * 4);

  const int TB = 256;
  // weight prep (fp32 -> fp16, (N,K) row-major)
  f32_to_f16_kernel<<<CDIV(2 * DIn * Cn, TB), TB, 0, stream>>>(inW[0], w_in0, 2 * DIn * Cn);
  f32_to_f16_kernel<<<CDIV(2 * DIn * Cn, TB), TB, 0, stream>>>(inW[1], w_in1, 2 * DIn * Cn);
  pad_xpw_kernel<<<CDIV(NPXP * DIn, TB), TB, 0, stream>>>(xpW[0], w_xp0);
  pad_xpw_kernel<<<CDIV(NPXP * DIn, TB), TB, 0, stream>>>(xpW[1], w_xp1);
  f32_to_f16_kernel<<<CDIV(Cn * DIn, TB), TB, 0, stream>>>(outW, w_out, Cn * DIn);
  f32_to_f16_kernel<<<CDIV(3 * HIDn * Cn, TB), TB, 0, stream>>>(msin, w_msin, 3 * HIDn * Cn);
  f32_to_f16_kernel<<<CDIV(Cn * HIDn, TB), TB, 0, stream>>>(msout, w_msout, Cn * HIDn);
  repack_conv_kernel<<<CDIV(9 * Cn * Cn, TB), TB, 0, stream>>>(c1W, w_c1);
  repack_conv_kernel<<<CDIV(9 * Cn * Cn, TB), TB, 0, stream>>>(c2W, w_c2);

  // ---- BiPixelMamba branch ----
  ln_cf_kernel<<<CDIV(Mn, TB), TB, 0, stream>>>(x, ln1_w, ln1_b, h32, xh);
  ln_row_kernel<<<CDIV(Mn, TB), TB, 0, stream>>>(h32, bmw, bmb, xn_h, 1e-5f);

  dim3 gIn(Mn / 64, (2 * DIn) / 64);
  gemm_wmma_kernel<4, Cn><<<gIn, 128, 0, stream>>>(xn_h, w_in0, xz0, 2 * DIn);
  gemm_wmma_kernel<4, Cn><<<gIn, 128, 0, stream>>>(xn_h, w_in1, xz1, 2 * DIn);

  dwconv_silu_kernel<<<CDIV(Mn * DIn, TB), TB, 0, stream>>>(xz0, convW[0], convB[0], uc0, 0);
  dwconv_silu_kernel<<<CDIV(Mn * DIn, TB), TB, 0, stream>>>(xz1, convW[1], convB[1], uc1, 1);

  dim3 gXp(Mn / 64, 1);
  gemm_wmma_kernel<3, DIn><<<gXp, 128, 0, stream>>>(uc0, w_xp0, db0, NPXP);
  gemm_wmma_kernel<3, DIn><<<gXp, 128, 0, stream>>>(uc1, w_xp1, db1, NPXP);

  delta_kernel<<<CDIV(Mn * DIn, TB), TB, 0, stream>>>(db0, dtW[0], dtB[0], de0);
  delta_kernel<<<CDIV(Mn * DIn, TB), TB, 0, stream>>>(db1, dtW[1], dtB[1], de1);

  scan_kernel<<<(Bn * DIn * DSn) / TB, TB, 0, stream>>>(de0, uc0, db0, Alog[0], ys0);
  scan_kernel<<<(Bn * DIn * DSn) / TB, TB, 0, stream>>>(de1, uc1, db1, Alog[1], ys1);

  combine_kernel<<<CDIV(Mn * DIn, TB), TB, 0, stream>>>(ys0, uc0, xz0, Dp[0],
                                                        ys1, uc1, xz1, Dp[1], ycomb);
  dim3 gOut(Mn / 64, Cn / 64);
  gemm_wmma_kernel<4, DIn><<<gOut, 128, 0, stream>>>(ycomb, w_out, bm32, Cn);
  residual1_kernel<<<CDIV(Mn * Cn, TB), TB, 0, stream>>>(x, bm32, h32, gamma1, xa32);

  // ---- MSFF branch ----
  ln_row_kernel<<<CDIV(Mn, TB), TB, 0, stream>>>(xa32, ln1_w, ln1_b, ln2h, 1e-6f);
  dim3 gMi(Mn / 64, (3 * HIDn) / 64);
  gemm_wmma_kernel<4, Cn><<<gMi, 128, 0, stream>>>(ln2h, w_msin, ph, 3 * HIDn);
  msff_kernel<<<CDIV(Mn * HIDn, TB), TB, 0, stream>>>(ph, dw1, dw2, dw3, g_h);
  dim3 gMo(Mn / 64, Cn / 64);
  gemm_wmma_kernel<4, HIDn><<<gMo, 128, 0, stream>>>(g_h, w_msout, gms, Cn);
  residual2_kernel<<<CDIV(Mn * Cn, TB), TB, 0, stream>>>(xa32, gms, gamma2, x2a);

  // ---- MultiScaleFusion on original input ----
  dim3 gCv(Bn * Hn * (Wn / 16) / 4, Cn / 64);
  conv3x3_wmma_kernel<<<gCv, 128, 0, stream>>>(xh, w_c1, c1B, c1o);
  bnrelu_kernel<<<CDIV(Mn * Cn, TB), TB, 0, stream>>>(c1o, bn1w, bn1b, bn1m, bn1v, f1h, f1f);
  conv3x3_wmma_kernel<<<gCv, 128, 0, stream>>>(f1h, w_c2, c2B, c2o);
  bnrelu_kernel<<<CDIV(Mn * Cn, TB), TB, 0, stream>>>(c2o, bn2w, bn2b, bn2m, bn2v, f2h, f2f);

  final_kernel<<<CDIV(Bn * Cn * Ln, TB), TB, 0, stream>>>(x2a, f2f, out);
}
